// DeformableMultiheadAttention_44495861187337
// MI455X (gfx1250) — matrix operand order repaired and checked by reading, not compiled, not run on hardware
//
#include <hip/hip_runtime.h>

// ---------------------------------------------------------------------------
// Deformable multi-head attention forward for MI455X (gfx1250, wave32, WMMA)
// B=8, DIM=384, H=W=32 (N=1024), HEADS=8 (d=48), GROUPS=4 (off_dim=96)
// ---------------------------------------------------------------------------

typedef __attribute__((ext_vector_type(16))) _Float16 v16h;
typedef __attribute__((ext_vector_type(8)))  _Float16 h8vec;
typedef __attribute__((ext_vector_type(8)))  float    v8f;

#define WMMA_F16(a, b, c) \
  __builtin_amdgcn_wmma_f32_16x16x32_f16(false, (a), false, (b), (short)0, (c), false, false)

// ---- WMMA fragment loaders -------------------------------------------------
// A-matrix 16x32 f16 (ISA dense-A table): lane half kOff=0/8,
//   halves 0..7 -> K = kOff+0..7, halves 8..15 -> K = kOff+16..23
__device__ __forceinline__ v16h ldA(const _Float16* p) {
  h8vec lo = *(const h8vec*)(p);
  h8vec hi = *(const h8vec*)(p + 16);
  v16h f;
#pragma unroll
  for (int i = 0; i < 8; ++i) { f[i] = lo[i]; f[8 + i] = hi[i]; }
  return f;
}
// B-matrix 32x16 f16 (same table order as A): lane half kOff=0/8,
//   halves 0..7 -> K = kOff+0..7, halves 8..15 -> K = kOff+16..23
__device__ __forceinline__ v16h ldB(const _Float16* p) {
  h8vec lo = *(const h8vec*)(p);
  h8vec hi = *(const h8vec*)(p + 16);
  v16h f;
#pragma unroll
  for (int i = 0; i < 8; ++i) { f[i] = lo[i]; f[8 + i] = hi[i]; }
  return f;
}

// ---- Conversion kernels ----------------------------------------------------
__global__ void cvt_weights(const float* __restrict__ Wq, const float* __restrict__ Wk,
                            const float* __restrict__ Wv, const float* __restrict__ Wp,
                            _Float16* __restrict__ o) {
  const int E = 384 * 384;
  int i = blockIdx.x * blockDim.x + threadIdx.x;
  if (i >= 4 * E) return;
  int m = i / E, j = i - m * E;
  const float* src = (m == 0) ? Wq : (m == 1) ? Wk : (m == 2) ? Wv : Wp;
  o[i] = (_Float16)src[j];
}

// x (B,C,N) fp32 -> xh (B*N, C) f16 row-major
__global__ void cvt_x(const float* __restrict__ x, _Float16* __restrict__ xh) {
  int i = blockIdx.x * blockDim.x + threadIdx.x;
  if (i >= 8 * 1024 * 384) return;
  int row = i / 384, col = i - row * 384;
  int b = row >> 10, n = row & 1023;
  xh[i] = (_Float16)x[(size_t)b * 384 * 1024 + (size_t)col * 1024 + n];
}

// ---- Generic WMMA GEMM: C = A(MxK) * W(NxK)^T ------------------------------
// MODE 0: q   -> outF fp32 (M,N) + outH f16 (M,N)
// MODE 1: k   -> outH f16 (M,N)
// MODE 2: v   -> outH f16 transposed per-head: Vt[(b*8+h)*48+d][n]
// MODE 3: prj -> outF fp32 scattered NCHW: out[b][col][n], + bias[col]
template <int MODE>
__global__ __launch_bounds__(128) void gemm_wmma(
    const _Float16* __restrict__ A, const _Float16* __restrict__ W,
    int M, int N, int K, float* __restrict__ outF, _Float16* __restrict__ outH,
    const float* __restrict__ bias) {
  int lane = threadIdx.x & 31;
  int wave = threadIdx.x >> 5;
  int mBase = blockIdx.x * 64 + (wave >> 1) * 32;
  int nBase = blockIdx.y * 64 + (wave & 1) * 32;

  int rsel = lane & 15;
  int aOff = (lane & 16) ? 8 : 0;
  int bOff = (lane & 16) ? 8 : 0;

  const _Float16* aRow0 = A + (size_t)(mBase + rsel) * K + aOff;
  const _Float16* aRow1 = A + (size_t)(mBase + 16 + rsel) * K + aOff;
  const _Float16* bCol0 = W + (size_t)(nBase + rsel) * K + bOff;
  const _Float16* bCol1 = W + (size_t)(nBase + 16 + rsel) * K + bOff;

  v8f acc[2][2] = {};
  for (int k = 0; k < K; k += 32) {
    __builtin_prefetch(aRow0 + k + 32, 0, 0);
    __builtin_prefetch(bCol0 + k + 32, 0, 0);
    v16h a0 = ldA(aRow0 + k);
    v16h a1 = ldA(aRow1 + k);
    v16h b0 = ldB(bCol0 + k);
    v16h b1 = ldB(bCol1 + k);
    acc[0][0] = WMMA_F16(a0, b0, acc[0][0]);
    acc[0][1] = WMMA_F16(a0, b1, acc[0][1]);
    acc[1][0] = WMMA_F16(a1, b0, acc[1][0]);
    acc[1][1] = WMMA_F16(a1, b1, acc[1][1]);
  }

  int rOff = (lane & 16) ? 8 : 0;
#pragma unroll
  for (int ti = 0; ti < 2; ++ti)
#pragma unroll
    for (int tj = 0; tj < 2; ++tj) {
      int col = nBase + tj * 16 + rsel;
#pragma unroll
      for (int r = 0; r < 8; ++r) {
        int row = mBase + ti * 16 + rOff + r;
        float v = acc[ti][tj][r];
        if (MODE == 0) {
          outF[(size_t)row * N + col] = v;
          outH[(size_t)row * N + col] = (_Float16)v;
        } else if (MODE == 1) {
          outH[(size_t)row * N + col] = (_Float16)v;
        } else if (MODE == 2) {
          int bb = row >> 10, n = row & 1023;
          int hh = col / 48, d = col - hh * 48;
          outH[(((size_t)(bb * 8 + hh) * 48 + d) << 10) + n] = (_Float16)v;
        } else {
          int bb = row >> 10, n = row & 1023;
          outF[(size_t)bb * 384 * 1024 + (size_t)col * 1024 + n] = v + bias[col];
        }
      }
    }
}

// ---- Offset generator: per-group 5x5 + 11x11 conv, tanh * 4 ----------------
// blockIdx.x = bg*2 + oc ; q fp32 (B*N, 384) ; off (32,2,1024)
__global__ __launch_bounds__(256) void offset_conv(
    const float* __restrict__ q, const float* __restrict__ W1,
    const float* __restrict__ W2, float* __restrict__ off) {
  __shared__ float w1s[96 * 25];
  __shared__ float w2s[96 * 121];
  int bg = blockIdx.x >> 1;
  int oc = blockIdx.x & 1;
  int b = bg >> 2, g = bg & 3;
  for (int i = threadIdx.x; i < 96 * 25; i += 256) w1s[i] = W1[oc * 96 * 25 + i];
  for (int i = threadIdx.x; i < 96 * 121; i += 256) w2s[i] = W2[oc * 96 * 121 + i];
  __syncthreads();
  const float* qb = q + (size_t)b * 1024 * 384 + g * 96;
  for (int p = threadIdx.x; p < 1024; p += 256) {
    int y = p >> 5, x = p & 31;
    float acc = 0.f;
    for (int c = 0; c < 96; ++c) {
      const float* w1r = &w1s[c * 25];
      const float* w2r = &w2s[c * 121];
#pragma unroll
      for (int ky = 0; ky < 5; ++ky) {
        int yy = y + ky - 2;
        if (yy < 0 || yy > 31) continue;
#pragma unroll
        for (int kx = 0; kx < 5; ++kx) {
          int xx = x + kx - 2;
          if (xx < 0 || xx > 31) continue;
          acc += qb[(size_t)(yy * 32 + xx) * 384 + c] * w1r[ky * 5 + kx];
        }
      }
      for (int ky = 0; ky < 11; ++ky) {
        int yy = y + ky - 5;
        if (yy < 0 || yy > 31) continue;
        for (int kx = 0; kx < 11; ++kx) {
          int xx = x + kx - 5;
          if (xx < 0 || xx > 31) continue;
          acc += qb[(size_t)(yy * 32 + xx) * 384 + c] * w2r[ky * 11 + kx];
        }
      }
    }
    off[((size_t)bg * 2 + oc) * 1024 + p] = tanhf(acc) * 4.0f;
  }
}

// ---- Bilinear gather -> kv f16 (B*N, 384) ---------------------------------
__global__ void bilinear_gather(const float* __restrict__ x,
                                const float* __restrict__ off,
                                _Float16* __restrict__ kv) {
  int idx = blockIdx.x * blockDim.x + threadIdx.x;
  if (idx >= 32 * 1024) return;
  int bg = idx >> 10, p = idx & 1023;
  int b = bg >> 2, g = bg & 3;
  int y = p >> 5, xq = p & 31;
  float oy = off[((size_t)bg * 2 + 0) * 1024 + p];
  float ox = off[((size_t)bg * 2 + 1) * 1024 + p];
  float ys = fminf(fmaxf((float)y + oy, 0.f), 31.f);
  float xs = fminf(fmaxf((float)xq + ox, 0.f), 31.f);
  float y0 = floorf(ys), x0 = floorf(xs);
  int y0i = (int)y0, x0i = (int)x0;
  int y1i = (y0i + 1 < 31) ? y0i + 1 : 31;
  int x1i = (x0i + 1 < 31) ? x0i + 1 : 31;
  float wy = ys - y0, wx = xs - x0;
  float w00 = (1.f - wy) * (1.f - wx), w01 = (1.f - wy) * wx;
  float w10 = wy * (1.f - wx), w11 = wy * wx;
  const float* xb = x + (size_t)b * 384 * 1024 + (size_t)g * 96 * 1024;
  int p00 = y0i * 32 + x0i, p01 = y0i * 32 + x1i;
  int p10 = y1i * 32 + x0i, p11 = y1i * 32 + x1i;
  _Float16* o = kv + ((size_t)b * 1024 + p) * 384 + g * 96;
  for (int c = 0; c < 96; ++c) {
    const float* xc = xb + (size_t)c * 1024;
    o[c] = (_Float16)(xc[p00] * w00 + xc[p01] * w01 + xc[p10] * w10 + xc[p11] * w11);
  }
}

// ---- Flash attention (transposed tiles) ------------------------------------
// Computes S^T = K Q^T and O^T = V^T P^T so that in the WMMA C-layout the
// QUERY index lives on the lane: the online-softmax max/sum reduce almost
// entirely in VALU registers with a single shfl_xor(16) per tile, and the
// running corr/1-over-l scaling of O^T is pure per-lane VALU (no broadcasts).
// grid (16 qtiles, 64 b*h), block 128 = 4 waves, 16 q-rows per wave.
__global__ __launch_bounds__(128) void flash_attn(
    const _Float16* __restrict__ Q, const _Float16* __restrict__ Km,
    const _Float16* __restrict__ Vt, const float* __restrict__ scale,
    _Float16* __restrict__ O) {
  __shared__ __align__(16) _Float16 pbuf[4][16][64];  // [wave][query][key]
  int lane = threadIdx.x & 31;
  int wave = threadIdx.x >> 5;
  int b = blockIdx.y >> 3, h = blockIdx.y & 7;
  size_t qGlob = (size_t)b * 1024 + blockIdx.x * 64 + wave * 16;

  float sc = scale[h];
  int rsel = lane & 15;
  int aOff = (lane & 16) ? 8 : 0;   // A-layout K interleave
  int bOff = (lane & 16) ? 8 : 0;   // B-layout K interleave (same as A)
  int rOff = (lane & 16) ? 8 : 0;   // C-layout row offset

  // Q fragments as B operand (col = query), d = 0..31 and 32..47 (pad to 64)
  const _Float16* qp = Q + (qGlob + rsel) * 384 + h * 48;
  v16h qb0, qb1 = {};
  {
    h8vec lo = *(const h8vec*)(qp + bOff);
    h8vec hi = *(const h8vec*)(qp + bOff + 16);
#pragma unroll
    for (int i = 0; i < 8; ++i) { qb0[i] = lo[i]; qb0[8 + i] = hi[i]; }
  }
  {  // d = 32+aOff..39+aOff in elements 0..7, elements 8..15 zero
    h8vec lo = *(const h8vec*)(qp + 32 + aOff);
#pragma unroll
    for (int i = 0; i < 8; ++i) { qb1[i] = lo[i]; }
  }

  float m = -1e30f, l = 0.f;  // per-lane stats: query = lane & 15
  v8f o[3] = {};              // O^T: lane = query, vgpr row = d within subtile

  for (int kt = 0; kt < 16; ++kt) {
    int keyBase = kt * 64;
    // ---- S^T = K Q^T : 4 key subtiles (keys on M / vgpr rows)
    v8f s[4] = {};
#pragma unroll
    for (int j = 0; j < 4; ++j) {
      const _Float16* kp =
          Km + ((size_t)b * 1024 + keyBase + j * 16 + rsel) * 384 + h * 48;
      v16h ka0, ka1 = {};
      {
        h8vec lo = *(const h8vec*)(kp + aOff);
        h8vec hi = *(const h8vec*)(kp + aOff + 16);
#pragma unroll
        for (int i = 0; i < 8; ++i) { ka0[i] = lo[i]; ka0[8 + i] = hi[i]; }
        h8vec lo1 = *(const h8vec*)(kp + 32 + aOff);  // d = 32+aOff..39+aOff < 48
#pragma unroll
        for (int i = 0; i < 8; ++i) { ka1[i] = lo1[i]; ka1[8 + i] = (_Float16)0.f; }
      }
      s[j] = WMMA_F16(ka0, qb0, s[j]);
      s[j] = WMMA_F16(ka1, qb1, s[j]);
    }
    // ---- online softmax: this lane holds 32 of the 64 keys for its query
    float mv = -1e30f;
#pragma unroll
    for (int j = 0; j < 4; ++j)
#pragma unroll
      for (int r = 0; r < 8; ++r) {
        s[j][r] *= sc;
        mv = fmaxf(mv, s[j][r]);
      }
    mv = fmaxf(mv, __shfl_xor(mv, 16));  // merge the partner half's 32 keys
    float mnew = fmaxf(m, mv);
    float corr = __expf(m - mnew);
    m = mnew;
    float ps = 0.f;
#pragma unroll
    for (int j = 0; j < 4; ++j)
#pragma unroll
      for (int r = 0; r < 8; ++r) {
        float pv = __expf(s[j][r] - mnew);
        s[j][r] = pv;
        ps += pv;
      }
    ps += __shfl_xor(ps, 16);
    l = l * corr + ps;
#pragma unroll
    for (int nt = 0; nt < 3; ++nt)
#pragma unroll
      for (int r = 0; r < 8; ++r) o[nt][r] *= corr;
    // ---- pack P^T into LDS rows [query][key] with 16B stores
#pragma unroll
    for (int j = 0; j < 4; ++j) {
      h8vec pk;
#pragma unroll
      for (int r = 0; r < 8; ++r) pk[r] = (_Float16)s[j][r];
      *(h8vec*)&pbuf[wave][rsel][j * 16 + rOff] = pk;
    }
    __syncthreads();
    // ---- P^T as B operand (col = query, contiguous keys)
    v16h pb0, pb1;
    {
      const _Float16* pp = &pbuf[wave][rsel][0];
      h8vec lo = *(const h8vec*)(pp + bOff);
      h8vec hi = *(const h8vec*)(pp + bOff + 16);
#pragma unroll
      for (int i = 0; i < 8; ++i) { pb0[i] = lo[i]; pb0[8 + i] = hi[i]; }
      h8vec lo1 = *(const h8vec*)(pp + 32 + bOff);
      h8vec hi1 = *(const h8vec*)(pp + 32 + bOff + 16);
#pragma unroll
      for (int i = 0; i < 8; ++i) { pb1[i] = lo1[i]; pb1[8 + i] = hi1[i]; }
    }
    // ---- O^T += V^T P^T  (Vt[(b*8+h)*48+d][n] is the A operand directly)
#pragma unroll
    for (int nt = 0; nt < 3; ++nt) {
      const _Float16* vp =
          Vt + ((size_t)(b * 8 + h) * 48 + nt * 16 + rsel) * 1024 + keyBase;
      v16h va0 = ldA(vp + aOff);
      v16h va1 = ldA(vp + 32 + aOff);
      o[nt] = WMMA_F16(va0, pb0, o[nt]);
      o[nt] = WMMA_F16(va1, pb1, o[nt]);
    }
    __syncthreads();
  }
  // ---- normalize (per-lane) and store f16 attention output (B*N, 384)
  float inv = 1.0f / l;
#pragma unroll
  for (int nt = 0; nt < 3; ++nt)
#pragma unroll
    for (int r = 0; r < 8; ++r)
      O[(qGlob + rsel) * 384 + h * 48 + nt * 16 + rOff + r] =
          (_Float16)(o[nt][r] * inv);
}

// ---------------------------------------------------------------------------
extern "C" void kernel_launch(void* const* d_in, const int* in_sizes, int n_in,
                              void* d_out, int out_size, void* d_ws, size_t ws_size,
                              hipStream_t stream) {
  (void)in_sizes; (void)n_in; (void)out_size; (void)ws_size;
  const float* x     = (const float*)d_in[0];
  const float* Wq    = (const float*)d_in[1];
  const float* Wk    = (const float*)d_in[2];
  const float* Wv    = (const float*)d_in[3];
  const float* Wp    = (const float*)d_in[4];
  const float* bproj = (const float*)d_in[5];
  const float* scale = (const float*)d_in[6];
  const float* W1    = (const float*)d_in[7];
  const float* W2    = (const float*)d_in[8];
  float* out = (float*)d_out;

  char* ws = (char*)d_ws;
  auto carve = [&](size_t bytes) -> char* {
    char* p = ws;
    ws += (bytes + 255) & ~(size_t)255;
    return p;
  };
  const size_t MN = (size_t)8192 * 384;
  _Float16* xh   = (_Float16*)carve(MN * 2);
  _Float16* wh   = (_Float16*)carve(4 * (size_t)147456 * 2);
  float*    qf   = (float*)   carve(MN * 4);
  _Float16* qh   = (_Float16*)carve(MN * 2);
  float*    offs = (float*)   carve((size_t)65536 * 4);
  _Float16* kvh  = (_Float16*)carve(MN * 2);
  _Float16* kh   = (_Float16*)carve(MN * 2);
  _Float16* vth  = (_Float16*)carve(MN * 2);
  _Float16* aoh  = (_Float16*)carve(MN * 2);

  cvt_weights<<<(4 * 147456 + 255) / 256, 256, 0, stream>>>(Wq, Wk, Wv, Wp, wh);
  cvt_x<<<(int)((MN + 255) / 256), 256, 0, stream>>>(x, xh);

  dim3 gg(8192 / 64, 384 / 64);
  // Q = X Wq^T (fp32 for conv + f16 for attention)
  gemm_wmma<0><<<gg, 128, 0, stream>>>(xh, wh, 8192, 384, 384, qf, qh, nullptr);
  // offsets = tanh(conv5(qg) + conv11(qg)) * 4
  offset_conv<<<64, 256, 0, stream>>>(qf, W1, W2, offs);
  // kv = bilinear(xg, grid + offsets)
  bilinear_gather<<<(32 * 1024 + 255) / 256, 256, 0, stream>>>(x, offs, kvh);
  // K, V projections
  gemm_wmma<1><<<gg, 128, 0, stream>>>(kvh, wh + 147456, 8192, 384, 384, nullptr, kh, nullptr);
  gemm_wmma<2><<<gg, 128, 0, stream>>>(kvh, wh + 2 * 147456, 8192, 384, 384, nullptr, vth, nullptr);
  // attention
  flash_attn<<<dim3(16, 64), 128, 0, stream>>>(qh, kh, vth, scale, aoh);
  // output projection + bias, scattered to (B,C,H,W)
  gemm_wmma<3><<<gg, 128, 0, stream>>>(aoh, wh + 3 * 147456, 8192, 384, 384, out, nullptr, bproj);
}
